// DenseContrastiveLossV2_18657337933965
// MI455X (gfx1250) — matrix operand order repaired and checked by reading, not compiled, not run on hardware
//
#include <hip/hip_runtime.h>

// ---------------- problem constants ----------------
#define NCLS      19          // real classes
#define NIMG      4
#define NCH       256         // C
#define NPIX      65536       // P = 256*256
#define NV        131         // V = 10000 / 76
#define NT        (NIMG*NCLS) // 76
#define M_TOTAL   (NT*NV)     // 9956 rows
#define MP        9984        // padded to 78*128 (rows >= M_TOTAL are zero)
#define ROWS_CLS  (4*NV)      // 524 same-class rows
#define POS_CNT   523.0f      // positives per row (4*131 - 1)

// GEMM tiling
#define BM  128
#define BK  32
#define LDA 40                // LDS row stride in halves (32 + 8 pad; 80B keeps 16B align, no bank conflicts)

typedef __attribute__((ext_vector_type(16))) _Float16 v16h;
typedef __attribute__((ext_vector_type(8)))  _Float16 v8h;
typedef __attribute__((ext_vector_type(8)))  float    v8f;

__device__ __forceinline__ int clsof(int r) { return (r / NV) % NCLS; }

// Row mapping for tile LOADS: never invalid — out-of-range rows map to the
// guaranteed-zero row M_TOTAL of fA, so async staging is divergence-free and
// every wave issues exactly 4 async ops per stage (deterministic ASYNCcnt).
__device__ __forceinline__ int map_row_load(int l, int pass, int kcls) {
  if (pass == 1) return l;                 // grid guarantees l < MP
  if (l >= ROWS_CLS) return M_TOTAL;       // zero row
  int b = l / NV;
  int off = l - b * NV;
  return (b * NCLS + kcls) * NV + off;
}

// Row mapping for the epilogue (loss accumulation): -1 = not a real row.
__device__ __forceinline__ int map_row_valid(int l, int pass, int kcls) {
  if (pass == 1) return (l < M_TOTAL) ? l : -1;
  if (l >= ROWS_CLS) return -1;
  int b = l / NV;
  int off = l - b * NV;
  return (b * NCLS + kcls) * NV + off;
}

// low 32 bits of a generic pointer to __shared__ == wave-relative LDS offset (ISA 10.2)
__device__ __forceinline__ unsigned lds_off(const void* p) {
  return (unsigned)(unsigned long long)p;
}

#define WAIT_ASYNC_4() asm volatile("s_wait_asynccnt 4" ::: "memory")
#define WAIT_ASYNC_0() asm volatile("s_wait_asynccnt 0" ::: "memory")

// ---------------- stage A: pick first V member pixels per (image,class) ----------------
__global__ void build_idx_kernel(const int* __restrict__ label, int* __restrict__ selIdx) {
  int t = blockIdx.x;              // 0..75  (t = n*19 + k)
  int n = t / NCLS, k = t % NCLS;
  int lane = threadIdx.x;          // wave32
  const int* lb = label + (size_t)n * (1024 * 1024);
  int count = 0;
  for (int p0 = 0; p0 < NPIX && count < NV; p0 += 32) {
    int p = p0 + lane;
    int i = p >> 8, j = p & 255;                   // dom pixel (i,j) -> label[4i][4j]
    int v = lb[(i << 2) * 1024 + (j << 2)];
    bool m = (v == k);
    unsigned mask = (unsigned)__ballot(m);
    int pre = __popc(mask & ((1u << lane) - 1u));
    if (m && (count + pre) < NV) selIdx[t * NV + count + pre] = p;
    count += __popc(mask);
  }
  if (lane == 0) {                                  // deterministic safety fill
    for (int q = count; q < NV; ++q) selIdx[t * NV + q] = 0;
  }
}

// ---------------- stage B: gather + L2-normalize + f16 convert ----------------
__global__ __launch_bounds__(256)
void gather_norm_kernel(const float* __restrict__ feat, const int* __restrict__ selIdx,
                        _Float16* __restrict__ fA) {
  int r = blockIdx.x;              // 0..MP-1
  int c = threadIdx.x;             // channel
  __shared__ float red[256];
  float x = 0.0f;
  if (r < M_TOTAL) {
    int t = r / NV, v = r - t * NV;
    int n = t / NCLS;
    int p = selIdx[t * NV + v];
    x = feat[(size_t)n * (NCH * (size_t)NPIX) + (size_t)c * NPIX + p];
  }
  red[c] = x * x;
  __syncthreads();
  for (int s = 128; s > 0; s >>= 1) { if (c < s) red[c] += red[c + s]; __syncthreads(); }
  float nrm = sqrtf(red[0]);
  float inv = 1.0f / fmaxf(nrm, 1e-12f);
  fA[(size_t)r * NCH + c] = (_Float16)(x * inv);   // rows >= M_TOTAL become zero
}

// ---------------- WMMA fragment loads (ISA 7.12.2 layouts) ----------------
__device__ __forceinline__ v16h load_frag_A(const _Float16* base) {
  int lane = threadIdx.x & 31;
  int m  = lane & 15;
  int kb = (lane < 16) ? 0 : 8;
  const _Float16* p = base + m * LDA;
  v8h lo = *(const v8h*)(p + kb);        // K = kb .. kb+7
  v8h hi = *(const v8h*)(p + kb + 16);   // K = kb+16 .. kb+23
  v16h r;
#pragma unroll
  for (int i = 0; i < 8; ++i) { r[i] = lo[i]; r[i + 8] = hi[i]; }
  return r;
}

__device__ __forceinline__ v16h load_frag_B(const _Float16* base) {
  int lane = threadIdx.x & 31;
  int nn = lane & 15;
  int kb = (lane < 16) ? 0 : 8;
  const _Float16* p = base + nn * LDA + kb;
  v8h lo = *(const v8h*)(p);             // K = kb .. kb+7
  v8h hi = *(const v8h*)(p + 16);        // K = kb+16 .. kb+23
  v16h r;
#pragma unroll
  for (int i = 0; i < 8; ++i) { r[i] = lo[i]; r[i + 8] = hi[i]; }
  return r;
}

// Stage one 128x32 A tile and one 128x32 B tile into LDS via the async
// global->LDS engine. 256 threads * 2 chunks each per tile; per WAVE this
// issues exactly 4 global_load_async_to_lds_b128 (counted on ASYNCcnt).
__device__ __forceinline__ void stage_tile_async(const _Float16* __restrict__ fA,
                                                 unsigned ldsA, unsigned ldsB,
                                                 int rowBase, int colBase, int k0,
                                                 int pass, int kcls, int tid) {
#pragma unroll
  for (int it = 0; it < 2; ++it) {
    int ch = tid + it * 256;              // 0..511
    int r = ch >> 2, part = ch & 3;       // row in tile, 16B chunk in row
    unsigned loff = (unsigned)((r * LDA + part * 8) * 2);
    int gA = map_row_load(rowBase + r, pass, kcls);
    int gB = map_row_load(colBase + r, pass, kcls);
    const _Float16* sa = fA + (size_t)gA * NCH + k0 + part * 8;
    const _Float16* sb = fA + (size_t)gB * NCH + k0 + part * 8;
    if (k0 + 2 * BK < NCH) __builtin_prefetch(sa + 2 * BK, 0, 3);  // global_prefetch_b8
    asm volatile("global_load_async_to_lds_b128 %0, %1, off"
                 :: "v"(ldsA + loff), "v"(sa) : "memory");
    asm volatile("global_load_async_to_lds_b128 %0, %1, off"
                 :: "v"(ldsB + loff), "v"(sb) : "memory");
  }
}

// ---------------- tiled Gram GEMM (pass 1: neg_sum, pass 2: pos_sum) ----------------
__global__ __launch_bounds__(256)
void gram_pass_kernel(const _Float16* __restrict__ fA, float* __restrict__ neg_sum,
                      float* __restrict__ pos_sum, int pass) {
  // pass1: symmetric -> only upper triangle of tiles
  if (pass == 1 && blockIdx.y > blockIdx.x) return;

  __shared__ __align__(16) _Float16 As[2][BM * LDA];
  __shared__ __align__(16) _Float16 Bs[2][BM * LDA];

  int kcls    = blockIdx.z;
  int rowBase = blockIdx.y * BM;
  int colBase = blockIdx.x * BM;
  int tid  = threadIdx.x;
  int lane = tid & 31;
  int wid  = tid >> 5;
  int wm   = wid & 3;   // wave row group: 32 rows
  int wn   = wid >> 2;  // wave col group: 64 cols

  unsigned ldsA[2] = { lds_off(&As[0][0]), lds_off(&As[1][0]) };
  unsigned ldsB[2] = { lds_off(&Bs[0][0]), lds_off(&Bs[1][0]) };

  v8f acc[2][4] = {};

  // prologue: stage k-slice 0 into buffer 0 (4 async ops outstanding per wave)
  stage_tile_async(fA, ldsA[0], ldsB[0], rowBase, colBase, 0, pass, kcls, tid);

#pragma unroll
  for (int ks = 0; ks < NCH / BK; ++ks) {            // 8 K-steps
    int k0  = ks * BK;
    int buf = ks & 1;
    if (ks < NCH / BK - 1) {
      // stage next slice into the other buffer -> 8 outstanding, then wait
      // until only the 4 newest remain (async ops complete in order).
      stage_tile_async(fA, ldsA[buf ^ 1], ldsB[buf ^ 1], rowBase, colBase,
                       k0 + BK, pass, kcls, tid);
      WAIT_ASYNC_4();
    } else {
      WAIT_ASYNC_0();
    }
    __syncthreads();                                 // all waves' tiles visible

    const _Float16* Ab = &As[buf][0];
    const _Float16* Bb = &Bs[buf][0];
    v16h a0 = load_frag_A(Ab + (wm * 32) * LDA);
    v16h a1 = load_frag_A(Ab + (wm * 32 + 16) * LDA);
#pragma unroll
    for (int nt = 0; nt < 4; ++nt) {
      v16h b = load_frag_B(Bb + (wn * 64 + nt * 16) * LDA);
      acc[0][nt] = __builtin_amdgcn_wmma_f32_16x16x32_f16(false, a0, false, b,
                                                          (short)0, acc[0][nt], false, false);
      acc[1][nt] = __builtin_amdgcn_wmma_f32_16x16x32_f16(false, a1, false, b,
                                                          (short)0, acc[1][nt], false, false);
    }
    __syncthreads();                                 // safe to overwrite this buffer
  }

  // ---------------- epilogue ----------------
  int nlocal  = lane & 15;
  int halfsel = (lane < 16) ? 0 : 8;   // C/D layout: VGPR i -> M=i (lanes 0-15) or M=8+i

  if (pass == 1) {
    bool offdiag = (rowBase != colBase);
    float colpart[4] = {0.f, 0.f, 0.f, 0.f};
    int gjv[4], cj[4];
#pragma unroll
    for (int nt = 0; nt < 4; ++nt) {
      int lcol = colBase + wn * 64 + nt * 16 + nlocal;
      gjv[nt] = (lcol < M_TOTAL) ? lcol : -1;
      cj[nt]  = (gjv[nt] >= 0) ? clsof(gjv[nt]) : -1;
    }
#pragma unroll
    for (int mt = 0; mt < 2; ++mt) {
#pragma unroll
      for (int i = 0; i < 8; ++i) {
        int lrow = rowBase + wm * 32 + mt * 16 + halfsel + i;
        int gi = (lrow < M_TOTAL) ? lrow : -1;
        int ci = (gi >= 0) ? clsof(gi) : -1;
        float rsum = 0.f;
#pragma unroll
        for (int nt = 0; nt < 4; ++nt) {
          float val = 0.f;
          if (gi >= 0 && gjv[nt] >= 0 && cj[nt] != ci)
            val = __expf(2.0f * acc[mt][nt][i]);       // logits = dot / 0.5
          rsum += val;
          if (offdiag) colpart[nt] += val;
        }
        rsum += __shfl_xor(rsum, 1, 32);
        rsum += __shfl_xor(rsum, 2, 32);
        rsum += __shfl_xor(rsum, 4, 32);
        rsum += __shfl_xor(rsum, 8, 32);
        if (nlocal == 0 && gi >= 0) atomicAdd(&neg_sum[gi], rsum);
      }
    }
    if (offdiag) {   // mirror contribution: columns become rows of the transposed tile
#pragma unroll
      for (int nt = 0; nt < 4; ++nt) {
        float v = colpart[nt] + __shfl_xor(colpart[nt], 16, 32);
        if (lane < 16 && gjv[nt] >= 0) atomicAdd(&neg_sum[gjv[nt]], v);
      }
    }
  } else {
#pragma unroll
    for (int mt = 0; mt < 2; ++mt) {
#pragma unroll
      for (int i = 0; i < 8; ++i) {
        int lrow = rowBase + wm * 32 + mt * 16 + halfsel + i;
        int gi = map_row_valid(lrow, 2, kcls);
        float ns = (gi >= 0) ? neg_sum[gi] : 0.f;
        float rsum = 0.f;
#pragma unroll
        for (int nt = 0; nt < 4; ++nt) {
          int lcol = colBase + wn * 64 + nt * 16 + nlocal;
          int gj = map_row_valid(lcol, 2, kcls);
          if (gi >= 0 && gj >= 0 && gi != gj) {
            float L = 2.0f * acc[mt][nt][i];
            rsum += L - __logf(__expf(L) + ns);        // log_prob
          }
        }
        rsum += __shfl_xor(rsum, 1, 32);
        rsum += __shfl_xor(rsum, 2, 32);
        rsum += __shfl_xor(rsum, 4, 32);
        rsum += __shfl_xor(rsum, 8, 32);
        if (nlocal == 0 && gi >= 0) atomicAdd(&pos_sum[gi], rsum);
      }
    }
  }
}

// ---------------- final reduction ----------------
__global__ __launch_bounds__(256)
void finalize_kernel(const float* __restrict__ pos_sum, float* __restrict__ out) {
  __shared__ float red[256];
  float s = 0.f;
  for (int i = threadIdx.x; i < M_TOTAL; i += 256) s += pos_sum[i];
  red[threadIdx.x] = s;
  __syncthreads();
  for (int st = 128; st > 0; st >>= 1) {
    if (threadIdx.x < st) red[threadIdx.x] += red[threadIdx.x + st];
    __syncthreads();
  }
  if (threadIdx.x == 0) out[0] = -red[0] / (POS_CNT * (float)M_TOTAL);
}

// ---------------- host launcher ----------------
extern "C" void kernel_launch(void* const* d_in, const int* in_sizes, int n_in,
                              void* d_out, int out_size, void* d_ws, size_t ws_size,
                              hipStream_t stream) {
  const int*   label = (const int*)d_in[0];     // [4,1024,1024] int32
  const float* feat  = (const float*)d_in[1];   // [4,256,256,256] f32
  float* out = (float*)d_out;

  char* ws = (char*)d_ws;
  int*      selIdx = (int*)ws;                          // 76*131*4 B
  size_t off = 40960;                                   // 16B aligned
  _Float16* fA = (_Float16*)(ws + off);                 // MP*256 halves = ~5.1 MB
  off += (size_t)MP * NCH * sizeof(_Float16);
  float* neg_sum = (float*)(ws + off);                  // MP floats
  float* pos_sum = (float*)(ws + off + (size_t)MP * sizeof(float));
  // zero both accumulators (contiguous) — graph-capture safe
  hipMemsetAsync(neg_sum, 0, (size_t)2 * MP * sizeof(float), stream);

  build_idx_kernel<<<NT, 32, 0, stream>>>(label, selIdx);
  gather_norm_kernel<<<MP, 256, 0, stream>>>(feat, selIdx, fA);
  gram_pass_kernel<<<dim3(MP / BM, MP / BM, 1), 256, 0, stream>>>(fA, neg_sum, pos_sum, 1);
  gram_pass_kernel<<<dim3(5, 5, NCLS), 256, 0, stream>>>(fA, neg_sum, pos_sum, 2);
  finalize_kernel<<<1, 256, 0, stream>>>(pos_sum, out);
}
